// DeepseekV2MLAAttention_78984448573828
// MI455X (gfx1250) — matrix operand order repaired and checked by reading, not compiled, not run on hardware
//
#include <hip/hip_runtime.h>
#include <math.h>

// ---------------- problem constants ----------------
#define T_      2048
#define HID_    2048
#define NH_     16
#define D_NOPE_ 128
#define D_ROPE_ 64
#define D_QK_   192      // D_NOPE + D_ROPE
#define D_V_    128
#define QLORA_  1536
#define KVLORA_ 512
#define NA_     2112     // QLORA + KVLORA + D_ROPE
#define NQB_    3072     // NH * D_QK
#define NKV_    4096     // NH * (D_NOPE + D_V)

typedef __attribute__((ext_vector_type(16))) __bf16 v16bf;
typedef __attribute__((ext_vector_type(8)))  float  v8f;

#define WMMA_BF16(a, b, c) \
    __builtin_amdgcn_wmma_f32_16x16x32_bf16(false, (a), false, (b), (short)0, (c), false, false)

// =====================================================================
// Packing helpers.
//
// A-fragment layout (16x32 bf16, wave32): lane L holds row m = L%16,
// half = L/16; element e (0..15) maps to k = ((e>>3)<<4) + (e&7) + 8*half.
// B-fragment layout (32x16 bf16): lane L holds col n = L%16, half = L/16;
// element e maps to k = ((e>>3)<<4) + (e&7) + 8*half, the same k order as A.
// Packed storage: tile-major [tile][lane][16 elems], 512 bf16 per tile.
//   A tiles: tile = mt*(K/32) + kt     B tiles: tile = kt*(N/16) + nt
// =====================================================================

__global__ void pack_a_kernel(const float* __restrict__ src, __bf16* __restrict__ dst,
                              int M, int K, int ld)
{
    size_t idx = (size_t)blockIdx.x * blockDim.x + threadIdx.x;
    size_t total = (size_t)M * K;
    if (idx >= total) return;
    int e    = (int)(idx & 15);
    int lane = (int)((idx >> 4) & 31);
    size_t tile = idx >> 9;
    int ktiles = K >> 5;
    int mt = (int)(tile / ktiles);
    int kt = (int)(tile % ktiles);
    int m = mt * 16 + (lane & 15);
    int k = kt * 32 + ((e >> 3) << 4) + (e & 7) + ((lane >> 4) << 3);
    dst[idx] = (__bf16)src[(size_t)m * ld + k];
}

__global__ void pack_b_kernel(const float* __restrict__ src, __bf16* __restrict__ dst,
                              int K, int N, int ld)
{
    size_t idx = (size_t)blockIdx.x * blockDim.x + threadIdx.x;
    size_t total = (size_t)K * N;
    if (idx >= total) return;
    int e    = (int)(idx & 15);
    int lane = (int)((idx >> 4) & 31);
    size_t tile = idx >> 9;
    int ntiles = N >> 4;
    int kt = (int)(tile / ntiles);
    int nt = (int)(tile % ntiles);
    int n = nt * 16 + (lane & 15);
    int k = kt * 32 + ((e >> 3) << 4) + (e & 7) + ((lane >> 4) << 3);
    dst[idx] = (__bf16)src[(size_t)k * ld + n];
}

// Build per-head packed K (B-layout, 192 x 2048): rows 0..127 = k_nope from kv,
// rows 128..191 = roped k_pe broadcast over heads.  grid.y = head.
__global__ void pack_b_khead_kernel(const float* __restrict__ kv,
                                    const float* __restrict__ kpe,
                                    __bf16* __restrict__ Kpack)
{
    int h = blockIdx.y;
    size_t idx = (size_t)blockIdx.x * blockDim.x + threadIdx.x;
    const size_t total = (size_t)D_QK_ * T_;
    if (idx >= total) return;
    int e    = (int)(idx & 15);
    int lane = (int)((idx >> 4) & 31);
    size_t tile = idx >> 9;
    const int ntiles = T_ >> 4;      // 128
    int kt = (int)(tile / ntiles);
    int nt = (int)(tile % ntiles);
    int s = nt * 16 + (lane & 15);   // key position (N dim)
    int d = kt * 32 + ((e >> 3) << 4) + (e & 7) + ((lane >> 4) << 3);  // feature (K dim)
    float v = (d < D_NOPE_)
                ? kv[(size_t)s * NKV_ + (size_t)h * (D_NOPE_ + D_V_) + d]
                : kpe[(size_t)s * D_ROPE_ + (d - D_NOPE_)];
    Kpack[(size_t)h * total + idx] = (__bf16)v;
}

// =====================================================================
// Packed bf16 WMMA GEMM: C[M,N] (fp32, ldc) = A(MxK) * B(KxN) * scale.
// Block = 128 threads (4 waves); block tile 128(M) x 64(N);
// wave tile 32 x 64 (2 A frags x 4 B frags = 8 WMMA per K-tile).
// Register double-buffered over K (ktiles always even here).
// grid = (N/64, M/128).  causal!=0 -> skip blocks strictly above diagonal.
// =====================================================================
__global__ __launch_bounds__(128)
void gemm_bf16_wmma(const __bf16* __restrict__ Ap, const __bf16* __restrict__ Bp,
                    float* __restrict__ C, int N, int K, int ldc,
                    float scale, int causal)
{
    const int lane = threadIdx.x & 31;
    const int wave = threadIdx.x >> 5;           // 0..3, each owns 32 rows
    const int bn = blockIdx.x * 64;
    const int bm = blockIdx.y * 128;
    if (causal && bn > bm + 127) return;         // fully-masked tile
    const int ktiles = K >> 5;
    const int ntiles = N >> 4;

    const int mt0 = (bm >> 4) + wave * 2;
    const __bf16* a0p = Ap + ((size_t)mt0 * ktiles) * 512 + lane * 16;
    const __bf16* a1p = a0p + (size_t)ktiles * 512;
    const __bf16* bp  = Bp + (size_t)(bn >> 4) * 512 + lane * 16;
    const size_t bstride = (size_t)ntiles * 512;

    v8f acc00 = {}, acc01 = {}, acc02 = {}, acc03 = {};
    v8f acc10 = {}, acc11 = {}, acc12 = {}, acc13 = {};

    // ---- prologue: load K-tile 0 into set A ----
    v16bf a0A = *(const v16bf*)(a0p);
    v16bf a1A = *(const v16bf*)(a1p);
    v16bf b0A = *(const v16bf*)(bp);
    v16bf b1A = *(const v16bf*)(bp + 512);
    v16bf b2A = *(const v16bf*)(bp + 1024);
    v16bf b3A = *(const v16bf*)(bp + 1536);

    for (int kt = 0; kt < ktiles; kt += 2) {
        // issue loads for K-tile kt+1 into set B
        const __bf16* bq = bp + bstride;
        v16bf a0B = *(const v16bf*)(a0p + 512);
        v16bf a1B = *(const v16bf*)(a1p + 512);
        v16bf b0B = *(const v16bf*)(bq);
        v16bf b1B = *(const v16bf*)(bq + 512);
        v16bf b2B = *(const v16bf*)(bq + 1024);
        v16bf b3B = *(const v16bf*)(bq + 1536);

        // compute on set A (loads issued one stage earlier)
        acc00 = WMMA_BF16(a0A, b0A, acc00);
        acc01 = WMMA_BF16(a0A, b1A, acc01);
        acc02 = WMMA_BF16(a0A, b2A, acc02);
        acc03 = WMMA_BF16(a0A, b3A, acc03);
        acc10 = WMMA_BF16(a1A, b0A, acc10);
        acc11 = WMMA_BF16(a1A, b1A, acc11);
        acc12 = WMMA_BF16(a1A, b2A, acc12);
        acc13 = WMMA_BF16(a1A, b3A, acc13);

        a0p += 1024;
        a1p += 1024;
        bp  += 2 * bstride;

        // issue loads for K-tile kt+2 into set A (if any)
        if (kt + 2 < ktiles) {
            a0A = *(const v16bf*)(a0p);
            a1A = *(const v16bf*)(a1p);
            b0A = *(const v16bf*)(bp);
            b1A = *(const v16bf*)(bp + 512);
            b2A = *(const v16bf*)(bp + 1024);
            b3A = *(const v16bf*)(bp + 1536);
        }

        // compute on set B
        acc00 = WMMA_BF16(a0B, b0B, acc00);
        acc01 = WMMA_BF16(a0B, b1B, acc01);
        acc02 = WMMA_BF16(a0B, b2B, acc02);
        acc03 = WMMA_BF16(a0B, b3B, acc03);
        acc10 = WMMA_BF16(a1B, b0B, acc10);
        acc11 = WMMA_BF16(a1B, b1B, acc11);
        acc12 = WMMA_BF16(a1B, b2B, acc12);
        acc13 = WMMA_BF16(a1B, b3B, acc13);
    }

    // C/D layout: lane holds col n = lane%16; vgpr r holds row r + 8*(lane/16)
    const int row0 = bm + wave * 32 + ((lane >> 4) << 3);
    const int col  = bn + (lane & 15);
#pragma unroll
    for (int r = 0; r < 8; ++r) {
        float* c0 = C + (size_t)(row0 + r) * ldc + col;
        c0[0]  = acc00[r] * scale;
        c0[16] = acc01[r] * scale;
        c0[32] = acc02[r] * scale;
        c0[48] = acc03[r] * scale;
        float* c1 = C + (size_t)(row0 + 16 + r) * ldc + col;
        c1[0]  = acc10[r] * scale;
        c1[16] = acc11[r] * scale;
        c1[32] = acc12[r] * scale;
        c1[48] = acc13[r] * scale;
    }
}

// =====================================================================
// RMSNorm of q_lat (1536) and kv_lat (512); writes bf16 directly in packed
// A-layout for the following WMMA GEMMs.  One block (256 thr) per row.
// =====================================================================
__global__ __launch_bounds__(256)
void rmsnorm_pack_kernel(const float* __restrict__ qkv_a,
                         const float* __restrict__ g_q,
                         const float* __restrict__ g_kv,
                         __bf16* __restrict__ qln,
                         __bf16* __restrict__ kvln)
{
    __shared__ float sm[8];
    const int t = blockIdx.x;
    const float* row = qkv_a + (size_t)t * NA_;

    // ---- q latent ----
    float ss = 0.f;
    for (int i = threadIdx.x; i < QLORA_; i += 256) { float v = row[i]; ss += v * v; }
    for (int off = 16; off; off >>= 1) ss += __shfl_xor(ss, off, 32);
    if ((threadIdx.x & 31) == 0) sm[threadIdx.x >> 5] = ss;
    __syncthreads();
    float tot = 0.f;
#pragma unroll
    for (int i = 0; i < 8; ++i) tot += sm[i];
    __syncthreads();
    float inv = rsqrtf(tot / (float)QLORA_ + 1e-6f);
    for (int i = threadIdx.x; i < QLORA_; i += 256) {
        float v = row[i] * inv * g_q[i];
        int kt = i >> 5, kk = i & 31;
        int half = (kk >> 3) & 1;
        int e = ((kk >> 4) << 3) + (kk & 7);
        size_t off_ = ((size_t)(t >> 4) * (QLORA_ >> 5) + kt) * 512
                    + (size_t)((t & 15) + 16 * half) * 16 + e;
        qln[off_] = (__bf16)v;
    }

    // ---- kv latent ----
    const float* rowkv = row + QLORA_;
    ss = 0.f;
    for (int i = threadIdx.x; i < KVLORA_; i += 256) { float v = rowkv[i]; ss += v * v; }
    for (int off = 16; off; off >>= 1) ss += __shfl_xor(ss, off, 32);
    if ((threadIdx.x & 31) == 0) sm[threadIdx.x >> 5] = ss;
    __syncthreads();
    tot = 0.f;
#pragma unroll
    for (int i = 0; i < 8; ++i) tot += sm[i];
    float inv2 = rsqrtf(tot / (float)KVLORA_ + 1e-6f);
    for (int i = threadIdx.x; i < KVLORA_; i += 256) {
        float v = rowkv[i] * inv2 * g_kv[i];
        int kt = i >> 5, kk = i & 31;
        int half = (kk >> 3) & 1;
        int e = ((kk >> 4) << 3) + (kk & 7);
        size_t off_ = ((size_t)(t >> 4) * (KVLORA_ >> 5) + kt) * 512
                    + (size_t)((t & 15) + 16 * half) * 16 + e;
        kvln[off_] = (__bf16)v;
    }
}

// ---------------- RoPE ----------------
__global__ void rope_kpe_kernel(const float* __restrict__ qkv_a,
                                const int* __restrict__ pos,
                                float* __restrict__ kpe)
{
    int idx = blockIdx.x * blockDim.x + threadIdx.x;
    if (idx >= T_ * 32) return;
    int i = idx & 31, t = idx >> 5;
    const float* src = qkv_a + (size_t)t * NA_ + QLORA_ + KVLORA_ + 2 * i;
    float x1 = src[0], x2 = src[1];
    float freq = __powf(10000.0f, -(float)(2 * i) / 64.0f);
    float ang = (float)pos[t] * freq;
    float c = cosf(ang), s = sinf(ang);
    kpe[(size_t)t * D_ROPE_ + 2 * i]     = x1 * c - x2 * s;
    kpe[(size_t)t * D_ROPE_ + 2 * i + 1] = x1 * s + x2 * c;
}

__global__ void rope_q_kernel(float* __restrict__ q, const int* __restrict__ pos)
{
    int idx = blockIdx.x * blockDim.x + threadIdx.x;
    if (idx >= T_ * NH_ * 32) return;
    int i = idx & 31, h = (idx >> 5) & (NH_ - 1), t = idx >> 9;
    float* ptr = q + (size_t)t * NQB_ + (size_t)h * D_QK_ + D_NOPE_ + 2 * i;
    float x1 = ptr[0], x2 = ptr[1];
    float freq = __powf(10000.0f, -(float)(2 * i) / 64.0f);
    float ang = (float)pos[t] * freq;
    float c = cosf(ang), s = sinf(ang);
    ptr[0] = x1 * c - x2 * s;
    ptr[1] = x1 * s + x2 * c;
}

// =====================================================================
// Causal softmax over S (fp32, already scaled), writing probs as bf16 in
// packed A-layout (zeros above the diagonal so the PV GEMM runs full K).
// One wave per row; 8 rows per 256-thread block.
// =====================================================================
__global__ __launch_bounds__(256)
void softmax_pack_kernel(const float* __restrict__ S, __bf16* __restrict__ Pp)
{
    int row  = blockIdx.x * 8 + (threadIdx.x >> 5);
    int lane = threadIdx.x & 31;
    const float* srow = S + (size_t)row * T_;
    const int limit = row;  // inclusive causal bound

    float mx = -3.4e38f;
    for (int s = lane; s <= limit; s += 32) mx = fmaxf(mx, srow[s]);
    for (int off = 16; off; off >>= 1) mx = fmaxf(mx, __shfl_xor(mx, off, 32));

    float sum = 0.f;
    for (int s = lane; s <= limit; s += 32) sum += __expf(srow[s] - mx);
    for (int off = 16; off; off >>= 1) sum += __shfl_xor(sum, off, 32);
    float rinv = 1.0f / sum;

    const int mt = row >> 4;
    const int lidx = row & 15;
    for (int s = lane; s < T_; s += 32) {
        float p = (s <= limit) ? __expf(srow[s] - mx) * rinv : 0.0f;
        int kt = s >> 5, kk = s & 31;
        int half = (kk >> 3) & 1;
        int e = ((kk >> 4) << 3) + (kk & 7);
        size_t off_ = ((size_t)mt * (T_ >> 5) + kt) * 512
                    + (size_t)(lidx + 16 * half) * 16 + e;
        Pp[off_] = (__bf16)p;
    }
}

// =====================================================================
// Launcher
// =====================================================================
extern "C" void kernel_launch(void* const* d_in, const int* in_sizes, int n_in,
                              void* d_out, int out_size, void* d_ws, size_t ws_size,
                              hipStream_t stream)
{
    (void)in_sizes; (void)n_in; (void)out_size; (void)ws_size;

    const float* hidden    = (const float*)d_in[0];
    const int*   positions = (const int*)  d_in[1];
    const float* w_qkv_a   = (const float*)d_in[2];
    const float* g_q       = (const float*)d_in[3];
    const float* w_q_b     = (const float*)d_in[4];
    const float* g_kv      = (const float*)d_in[5];
    const float* w_kv_b    = (const float*)d_in[6];
    const float* w_o       = (const float*)d_in[7];
    float* out = (float*)d_out;

    char* p = (char*)d_ws;
    auto carve = [&](size_t bytes) -> void* {
        void* r = (void*)p;
        p += (bytes + 255) & ~(size_t)255;
        return r;
    };

    __bf16* Hpack    = (__bf16*)carve((size_t)T_ * HID_   * 2);  // hidden, A-packed
    __bf16* WaPack   = (__bf16*)carve((size_t)HID_ * NA_  * 2);  // w_qkv_a, B-packed
    __bf16* WqbPack  = (__bf16*)carve((size_t)QLORA_ * NQB_ * 2);
    __bf16* WkvbPack = (__bf16*)carve((size_t)KVLORA_ * NKV_ * 2);
    __bf16* WoPack   = (__bf16*)carve((size_t)HID_ * HID_ * 2);
    float*  qkv_a    = (float*) carve((size_t)T_ * NA_    * 4);
    __bf16* qln      = (__bf16*)carve((size_t)T_ * QLORA_ * 2);  // A-packed
    __bf16* kvln     = (__bf16*)carve((size_t)T_ * KVLORA_* 2);  // A-packed
    float*  kpe      = (float*) carve((size_t)T_ * D_ROPE_* 4);
    float*  qbuf     = (float*) carve((size_t)T_ * NQB_   * 4);
    float*  kvbuf    = (float*) carve((size_t)T_ * NKV_   * 4);
    __bf16* Qpack    = (__bf16*)carve((size_t)NH_ * T_ * D_QK_ * 2);
    __bf16* Kpack    = (__bf16*)carve((size_t)NH_ * D_QK_ * T_ * 2);
    __bf16* Vpack    = (__bf16*)carve((size_t)NH_ * T_ * D_V_  * 2);
    float*  Sbuf     = (float*) carve((size_t)T_ * T_     * 4);  // reused per head
    __bf16* Ppack    = (__bf16*)carve((size_t)T_ * T_     * 2);  // reused per head
    float*  attn     = (float*) carve((size_t)T_ * HID_   * 4);
    __bf16* AttnPack = (__bf16*)carve((size_t)T_ * HID_   * 2);

    const int TB = 256;
    auto nblk = [](size_t n) { return (unsigned)((n + 255) / 256); };

    // ---- weight / activation packing ----
    pack_a_kernel<<<nblk((size_t)T_ * HID_),    TB, 0, stream>>>(hidden,  Hpack,   T_, HID_, HID_);
    pack_b_kernel<<<nblk((size_t)HID_ * NA_),   TB, 0, stream>>>(w_qkv_a, WaPack,  HID_, NA_, NA_);
    pack_b_kernel<<<nblk((size_t)QLORA_*NQB_),  TB, 0, stream>>>(w_q_b,   WqbPack, QLORA_, NQB_, NQB_);
    pack_b_kernel<<<nblk((size_t)KVLORA_*NKV_), TB, 0, stream>>>(w_kv_b,  WkvbPack,KVLORA_, NKV_, NKV_);
    pack_b_kernel<<<nblk((size_t)HID_ * HID_),  TB, 0, stream>>>(w_o,     WoPack,  HID_, HID_, HID_);

    // ---- qkv_a = hidden @ w_qkv_a ----
    gemm_bf16_wmma<<<dim3(NA_/64, T_/128), 128, 0, stream>>>(Hpack, WaPack, qkv_a, NA_, HID_, NA_, 1.0f, 0);

    // ---- rmsnorm (packed bf16 out) + k_pe rope ----
    rmsnorm_pack_kernel<<<T_, 256, 0, stream>>>(qkv_a, g_q, g_kv, qln, kvln);
    rope_kpe_kernel<<<nblk((size_t)T_ * 32), TB, 0, stream>>>(qkv_a, positions, kpe);

    // ---- q = qln @ w_q_b ; kv = kvln @ w_kv_b ----
    gemm_bf16_wmma<<<dim3(NQB_/64, T_/128), 128, 0, stream>>>(qln,  WqbPack,  qbuf,  NQB_, QLORA_, NQB_, 1.0f, 0);
    gemm_bf16_wmma<<<dim3(NKV_/64, T_/128), 128, 0, stream>>>(kvln, WkvbPack, kvbuf, NKV_, KVLORA_, NKV_, 1.0f, 0);

    // ---- rope q_pe in place ----
    rope_q_kernel<<<nblk((size_t)T_ * NH_ * 32), TB, 0, stream>>>(qbuf, positions);

    // ---- per-head operand packing ----
    for (int h = 0; h < NH_; ++h) {
        pack_a_kernel<<<nblk((size_t)T_ * D_QK_), TB, 0, stream>>>(
            qbuf + (size_t)h * D_QK_, Qpack + (size_t)h * T_ * D_QK_, T_, D_QK_, NQB_);
        pack_b_kernel<<<nblk((size_t)T_ * D_V_), TB, 0, stream>>>(
            kvbuf + (size_t)h * (D_NOPE_ + D_V_) + D_NOPE_,
            Vpack + (size_t)h * T_ * D_V_, T_, D_V_, NKV_);
    }
    pack_b_khead_kernel<<<dim3(nblk((size_t)D_QK_ * T_), NH_), TB, 0, stream>>>(kvbuf, kpe, Kpack);

    // ---- attention, head by head (stream-ordered; S/Ppack reused) ----
    const float scaling = 1.0f / sqrtf((float)D_QK_);
    for (int h = 0; h < NH_; ++h) {
        const __bf16* Qh = Qpack + (size_t)h * T_ * D_QK_;
        const __bf16* Kh = Kpack + (size_t)h * D_QK_ * T_;
        const __bf16* Vh = Vpack + (size_t)h * T_ * D_V_;
        gemm_bf16_wmma<<<dim3(T_/64, T_/128), 128, 0, stream>>>(Qh, Kh, Sbuf, T_, D_QK_, T_, scaling, 1);
        softmax_pack_kernel<<<T_/8, 256, 0, stream>>>(Sbuf, Ppack);
        gemm_bf16_wmma<<<dim3(D_V_/64, T_/128), 128, 0, stream>>>(Ppack, Vh, attn + (size_t)h * D_V_, D_V_, T_, HID_, 1.0f, 0);
    }

    // ---- out = attn @ w_o ----
    pack_a_kernel<<<nblk((size_t)T_ * HID_), TB, 0, stream>>>(attn, AttnPack, T_, HID_, HID_);
    gemm_bf16_wmma<<<dim3(HID_/64, T_/128), 128, 0, stream>>>(AttnPack, WoPack, out, HID_, HID_, HID_, 1.0f, 0);
}
